// ManifoldLinear_53996328846020
// MI455X (gfx1250) — matrix operand order repaired and checked by reading, not compiled, not run on hardware
//
#include <hip/hip_runtime.h>
#include <math.h>

#define TOKENS     256
#define IN_F       4096
#define OUT_F      4096
#define NUM_CHUNKS 16384
#define CHUNK_SIZE 1024
#define D_ALPHA    64
#define HIDDEN     256

typedef __attribute__((ext_vector_type(16))) __bf16 v16bf;
typedef __attribute__((ext_vector_type(8)))  float  v8f;

union V16 { v16bf v; uint4 q[2]; };

__device__ __forceinline__ unsigned short bf16_rne(float f) {
  unsigned u = __float_as_uint(f);
  u += 0x7FFFu + ((u >> 16) & 1u);
  return (unsigned short)(u >> 16);
}
__device__ __forceinline__ float bf16_f32(unsigned short h) {
  return __uint_as_float(((unsigned)h) << 16);
}
__device__ __forceinline__ void bf16_split(float v, unsigned short* hi, unsigned short* lo) {
  unsigned short h = bf16_rne(v);
  *hi = h;
  *lo = bf16_rne(v - bf16_f32(h));
}

__device__ __forceinline__ v8f wmma_bf16(v16bf a, v16bf b, v8f c) {
  // v_wmma_f32_16x16x32_bf16: D = A(16x32) * B(32x16) + C, f32 accumulate
  return __builtin_amdgcn_wmma_f32_16x16x32_bf16(false, a, false, b, (short)0, c, false, false);
}

// CDNA5 async copy: 16B per lane from global saddr+vaddr into LDS at vdst, tracked by ASYNCcnt
__device__ __forceinline__ void async_ld_b128(unsigned lds_off, unsigned goff, const void* base) {
  asm volatile("global_load_async_to_lds_b128 %0, %1, %2"
               :: "v"(lds_off), "v"(goff), "s"(base) : "memory");
}
__device__ __forceinline__ void wait_async0() {
  asm volatile("s_wait_asynccnt 0" ::: "memory");
}

// ---------------------------------------------------------------------------
// Kernel 1: split x and w2 into bf16 hi/lo planes
// ---------------------------------------------------------------------------
__global__ __launch_bounds__(256) void ml_prep(const float* __restrict__ x,
                                               const float* __restrict__ w2,
                                               unsigned short* __restrict__ xh,
                                               unsigned short* __restrict__ xl,
                                               unsigned short* __restrict__ w2h,
                                               unsigned short* __restrict__ w2l) {
  int i = blockIdx.x * 256 + threadIdx.x;
  const int NX  = TOKENS * IN_F;         // 1048576
  const int NW2 = CHUNK_SIZE * HIDDEN;   // 262144
  if (i < NX) {
    bf16_split(x[i], &xh[i], &xl[i]);
  } else if (i < NX + NW2) {
    int j = i - NX;
    bf16_split(w2[j], &w2h[j], &w2l[j]);
  }
}

// ---------------------------------------------------------------------------
// Kernel 2: decoder. Block = 16 chunks. GEMM1+GeLU in exact f32 VALU,
// GEMM2 (16x1024, K=256) via bf16x3 WMMA with 3 independent acc chains.
// ---------------------------------------------------------------------------
__global__ __launch_bounds__(256) void ml_decoder(const float* __restrict__ codebook,
                                                  const float* __restrict__ w1,
                                                  const float* __restrict__ b1,
                                                  const unsigned short* __restrict__ w2h,
                                                  const unsigned short* __restrict__ w2l,
                                                  const float* __restrict__ b2,
                                                  unsigned short* __restrict__ wh,
                                                  unsigned short* __restrict__ wl) {
  __shared__ __align__(16) float          cb[16 * 64];       // 4KB codebook tile
  __shared__ __align__(16) unsigned short hh[16 * HIDDEN];   // 8KB h hi
  __shared__ __align__(16) unsigned short hl[16 * HIDDEN];   // 8KB h lo

  const int t  = threadIdx.x;
  const int c0 = blockIdx.x * 16;

  // stage codebook rows (16 x 64 f32) into LDS
  ((float4*)cb)[t] = ((const float4*)(codebook + (size_t)c0 * D_ALPHA))[t];
  __syncthreads();

  // GEMM1 + exact GeLU: each thread -> one chunk row m, 16 hidden units
  {
    const int m = t >> 4;
    const int rbase = (t & 15) * 16;
    for (int ii = 0; ii < 16; ++ii) {
      const int r = rbase + ii;
      const float* wrow = w1 + r * D_ALPHA;
      float s = b1[r];
#pragma unroll
      for (int k = 0; k < D_ALPHA; ++k) s += cb[m * D_ALPHA + k] * wrow[k];
      const float g = 0.5f * s * (1.0f + erff(s * 0.70710678118654752f));
      bf16_split(g, &hh[m * HIDDEN + r], &hl[m * HIDDEN + r]);
    }
  }
  __syncthreads();

  // GEMM2: chunks[16 x 1024] = h(16x256) @ w2^T, bf16x3 WMMA
  const int lane = t & 31;
  const int wv   = t >> 5;        // 8 waves, each owns 128 output cols
  const int lhi  = lane >> 4;     // 0/1 half-wave
  const int lm   = lane & 15;

  for (int nt = 0; nt < 8; ++nt) {
    const int jb = wv * 128 + nt * 16;
    v8f acc0 = {}, acc1 = {}, acc2 = {};   // independent chains
    for (int k0 = 0; k0 < HIDDEN; k0 += 32) {
      // A tile (h) from LDS: lane lm = row M, halves per ISA A-layout
      V16 ah, al;
      const unsigned short* pa = hh + lm * HIDDEN + k0 + 8 * lhi;
      ah.q[0] = *(const uint4*)(pa);
      ah.q[1] = *(const uint4*)(pa + 16);
      const unsigned short* pal = hl + lm * HIDDEN + k0 + 8 * lhi;
      al.q[0] = *(const uint4*)(pal);
      al.q[1] = *(const uint4*)(pal + 16);
      // B tile: B[k][n] = w2[jb+n][k]; lane lm = column n, per ISA B-layout
      V16 bh, bl;
      const unsigned short* pb = w2h + (size_t)(jb + lm) * HIDDEN + k0 + 8 * lhi;
      bh.q[0] = *(const uint4*)(pb);
      bh.q[1] = *(const uint4*)(pb + 16);
      const unsigned short* pbl = w2l + (size_t)(jb + lm) * HIDDEN + k0 + 8 * lhi;
      bl.q[0] = *(const uint4*)(pbl);
      bl.q[1] = *(const uint4*)(pbl + 16);

      acc0 = wmma_bf16(ah.v, bh.v, acc0);
      acc1 = wmma_bf16(ah.v, bl.v, acc1);
      acc2 = wmma_bf16(al.v, bh.v, acc2);
    }
    // epilogue: combine chains, add b2, split to hi/lo, store flat
    const float bb = b2[jb + lm];
#pragma unroll
    for (int v = 0; v < 8; ++v) {
      const int    mm   = v + 8 * lhi;             // C/D layout: M = v (+8 upper half)
      const size_t flat = (size_t)(c0 + mm) * CHUNK_SIZE + (jb + lm);
      const float  val  = acc0[v] + acc1[v] + acc2[v] + bb;
      unsigned short hi, lo;
      bf16_split(val, &hi, &lo);
      wh[flat] = hi;
      wl[flat] = lo;
    }
  }
}

// ---------------------------------------------------------------------------
// Kernel 3: out(256x4096) = x @ w^T + bias, bf16x3 WMMA, K=4096.
// Block tile M=32 x N=128, 8 waves (2 M-groups x 4 N-groups), wave = 16x32.
// x k-slices double-buffered in LDS via CDNA5 async-to-LDS loads:
// one barrier per k-step, prefetch of k+1 overlaps WMMA compute of k.
// ---------------------------------------------------------------------------
__global__ __launch_bounds__(256) void ml_gemm_out(const unsigned short* __restrict__ xplanes,
                                                   const unsigned short* __restrict__ wh,
                                                   const unsigned short* __restrict__ wl,
                                                   const float* __restrict__ bias,
                                                   float* __restrict__ out) {
  // two buffers, each: hi plane (32x32 bf16, 2KB) + lo plane (2KB)
  __shared__ __align__(16) unsigned short atile[2 * 2048];

  const int t      = threadIdx.x;
  const int lane   = t & 31;
  const int wv     = t >> 5;
  const int mgroup = wv >> 2;            // 0..1
  const int ngroup = wv & 3;             // 0..3
  const int lhi    = lane >> 4;
  const int lm     = lane & 15;

  const int m0    = blockIdx.y * 32;     // token rows
  const int obase = blockIdx.x * 128;    // output feature cols

  // per-thread async staging address pieces: 256 threads * 16B = 4KB tile
  const int p     = t >> 7;              // 0 = hi plane, 1 = lo plane
  const int rr    = (t & 127) >> 2;      // row within 32-row tile
  const int cbyte = (t & 3) * 16;        // 16B segment within 64B row
  const unsigned ldsBase = (unsigned)(size_t)(&atile[0]);
  const unsigned goffRow = (unsigned)p * (TOKENS * IN_F * 2u)
                         + (unsigned)(m0 + rr) * (IN_F * 2u) + (unsigned)cbyte;

  v8f acc[2][3];
#pragma unroll
  for (int nt = 0; nt < 2; ++nt)
#pragma unroll
    for (int c = 0; c < 3; ++c) acc[nt][c] = (v8f){};

  const int NK = IN_F / 32;              // 128 k-steps

  // prime the pipeline: prefetch k-step 0 into buffer 0
  async_ld_b128(ldsBase + (unsigned)t * 16u, goffRow, xplanes);

  int cur = 0;
  for (int ki = 0; ki < NK; ++ki) {
    const int k0 = ki * 32;
    wait_async0();       // my part of buffer `cur` landed
    __syncthreads();     // everyone's part landed; everyone done reading cur^1

    if (ki + 1 < NK) {   // prefetch next k-slice into the other buffer
      async_ld_b128(ldsBase + (unsigned)(cur ^ 1) * 4096u + (unsigned)t * 16u,
                    goffRow + (unsigned)(k0 + 32) * 2u, xplanes);
    }

    // A tile (x) from LDS buffer `cur`, 16x32 per wave
    V16 ah, al;
    const unsigned short* pa = atile + cur * 2048 + (mgroup * 16 + lm) * 32 + 8 * lhi;
    ah.q[0] = *(const uint4*)(pa);
    ah.q[1] = *(const uint4*)(pa + 16);
    const unsigned short* pal = pa + 1024;
    al.q[0] = *(const uint4*)(pal);
    al.q[1] = *(const uint4*)(pal + 16);

#pragma unroll
    for (int nt = 0; nt < 2; ++nt) {
      const int orow = obase + ngroup * 32 + nt * 16 + lm;
      V16 bh, bl;
      const unsigned short* pb = wh + (size_t)orow * IN_F + k0 + 8 * lhi;
      bh.q[0] = *(const uint4*)(pb);
      bh.q[1] = *(const uint4*)(pb + 16);
      const unsigned short* pbl = wl + (size_t)orow * IN_F + k0 + 8 * lhi;
      bl.q[0] = *(const uint4*)(pbl);
      bl.q[1] = *(const uint4*)(pbl + 16);

      acc[nt][0] = wmma_bf16(ah.v, bh.v, acc[nt][0]);
      acc[nt][1] = wmma_bf16(ah.v, bl.v, acc[nt][1]);
      acc[nt][2] = wmma_bf16(al.v, bh.v, acc[nt][2]);
    }
    cur ^= 1;
  }

  // epilogue: combine chains, add bias, store f32
#pragma unroll
  for (int nt = 0; nt < 2; ++nt) {
    const int on = obase + ngroup * 32 + nt * 16 + lm;
    const float bb = bias[on];
#pragma unroll
    for (int v = 0; v < 8; ++v) {
      const int tok = m0 + mgroup * 16 + v + 8 * lhi;
      out[(size_t)tok * OUT_F + on] = acc[nt][0][v] + acc[nt][1][v] + acc[nt][2][v] + bb;
    }
  }
}

// ---------------------------------------------------------------------------
extern "C" void kernel_launch(void* const* d_in, const int* in_sizes, int n_in,
                              void* d_out, int out_size, void* d_ws, size_t ws_size,
                              hipStream_t stream) {
  const float* x        = (const float*)d_in[0];
  const float* codebook = (const float*)d_in[1];
  const float* w1       = (const float*)d_in[2];
  const float* b1       = (const float*)d_in[3];
  const float* w2       = (const float*)d_in[4];
  const float* b2       = (const float*)d_in[5];
  const float* bias     = (const float*)d_in[6];
  float* out = (float*)d_out;

  char* ws = (char*)d_ws;
  // workspace layout:
  //   [0, 4MB)  : x hi plane then x lo plane (each TOKENS*IN_F bf16)
  //   [4, 5MB)  : w2 hi plane then w2 lo plane
  //   [5, 69MB) : w hi plane (32MB) then w lo plane (32MB)
  unsigned short* xplanes = (unsigned short*)(ws);
  unsigned short* xh = xplanes;
  unsigned short* xl = xplanes + (size_t)TOKENS * IN_F;
  unsigned short* w2h = (unsigned short*)(ws + (4ull << 20));
  unsigned short* w2l = w2h + (size_t)CHUNK_SIZE * HIDDEN;
  unsigned short* whp = (unsigned short*)(ws + (5ull << 20));
  unsigned short* wlp = whp + (size_t)OUT_F * IN_F;

  ml_prep<<<5120, 256, 0, stream>>>(x, w2, xh, xl, w2h, w2l);
  ml_decoder<<<NUM_CHUNKS / 16, 256, 0, stream>>>(codebook, w1, b1, w2h, w2l, b2, whp, wlp);
  ml_gemm_out<<<dim3(OUT_F / 128, TOKENS / 32), 256, 0, stream>>>(xplanes, whp, wlp, bias, out);
}
